// GatedGN_32238024524181
// MI455X (gfx1250) — matrix operand order repaired and checked by reading, not compiled, not run on hardware
//
#include <hip/hip_runtime.h>
#include <hip/hip_bf16.h>
#include <math.h>

// ----------------------------------------------------------------------------
// GatedGN (N=2048, K=8, NIN=14, H=128, L=2, k=3) for MI455X / gfx1250.
//
// Roofline: ~210 GFLOP of GEMM work but ~170 MB of edge-activation traffic per
// message-passing iteration -> bandwidth-bound at 23.3 TB/s. Therefore:
//   * all activations / hidden states are bf16 (halves HBM bytes vs fp32)
//   * GEMMs use v_wmma_f32_16x16x32_bf16 (f32 accumulate for gate numerics)
//   * gi = x@Wih^T and gh = h@Whh^T plus bias + sigmoid/tanh gate math are
//     fused in ONE kernel; gi/gh (16x384 f32) live only in LDS.
// Weights are converted to padded bf16 once per call (deterministic).
// Workspace requirement: ~194 MB.
// ----------------------------------------------------------------------------

typedef __bf16 bf16;
typedef __attribute__((ext_vector_type(16))) __bf16 v16bf;
typedef __attribute__((ext_vector_type(8)))  float  v8f;

#define HDIM 128
#define G3   384   // 3*H
#define NE   131072ul  // N*K*K edge rows
#define NN   16384ul   // N*K node rows
#define NG   2048ul    // N global rows

// ---------------------------------------------------------------------------
// Fused GRU cell for a 16-row tile. 256 threads = 8 waves (wave32).
//   X   : [rows][Kx] bf16, Kx % 32 == 0 (zero-padded input)
//   Hst : [rows][128] bf16 hidden, updated in place
//   Wih : [384][Kx] bf16 (padded), Whh : [384][128] bf16
//   Xout: optional bf16 sink for the new hidden (feeds next layer's x)
// ---------------------------------------------------------------------------
__global__ __launch_bounds__(256) void gru16_kernel(
    const bf16* __restrict__ X, int Kx,
    bf16* __restrict__ Hst,
    const bf16* __restrict__ Wih,
    const bf16* __restrict__ Whh,
    const float* __restrict__ bih,
    const float* __restrict__ bhh,
    bf16* __restrict__ Xout, int xld, int xoff)
{
    __shared__ bf16  sX[16 * 256];
    __shared__ bf16  sH[16 * HDIM];
    __shared__ float sGi[16 * G3];
    __shared__ float sGh[16 * G3];

    const int    tid  = threadIdx.x;
    const size_t row0 = (size_t)blockIdx.x * 16;

    // ---- phase 1: stage x and h tiles into LDS (bf16) ----
    for (int idx = tid; idx < 16 * Kx; idx += 256) {
        int r = idx / Kx, c = idx - r * Kx;
        sX[r * Kx + c] = X[(row0 + r) * (size_t)Kx + c];
    }
    for (int idx = tid; idx < 16 * HDIM; idx += 256)
        sH[idx] = Hst[row0 * HDIM + idx];
    __syncthreads();

    // ---- phase 2: 48 WMMA tiles (24 gi + 24 gh), 6 per wave ----
    const int wave = tid >> 5, lane = tid & 31;
    const int n16 = lane & 15, hi = lane >> 4;

    for (int t = wave; t < 48; t += 8) {            // wave-uniform loop
        const bool  isGi = (t < 24);
        const int   ct   = isGi ? t : t - 24;       // 16-col tile of 384
        const bf16* W    = isGi ? Wih : Whh;
        const bf16* A    = isGi ? sX  : sH;
        const int   Kd   = isGi ? Kx  : HDIM;

        v8f acc = {};
        for (int k0 = 0; k0 < Kd; k0 += 32) {
            // A fragment: 16x32 bf16; lanes 0-15 hold K = k0+0..15,
            // lanes 16-31 hold K = k0+16..31 (matches B's k order)
            v16bf a;
            const bf16* arow = A + n16 * Kd + k0 + hi * 16;
            #pragma unroll
            for (int e = 0; e < 8; ++e) { a[e] = arow[e]; a[8 + e] = arow[8 + e]; }
            // B fragment: 32x16, B[k][n] = W[ct*16+n][k]; lo lanes K=k0..k0+15,
            // hi lanes K=k0+16..k0+31 -> 32B contiguous load per lane
            v16bf b;
            const bf16* wrow = W + (size_t)(ct * 16 + n16) * Kd + k0 + hi * 16;
            #pragma unroll
            for (int e = 0; e < 16; ++e) b[e] = wrow[e];
            acc = __builtin_amdgcn_wmma_f32_16x16x32_bf16(
                false, a, false, b, (short)0, acc, false, false);
        }
        float* G = isGi ? sGi : sGh;
        #pragma unroll
        for (int v = 0; v < 8; ++v)                 // D: VGPR v <-> M = v + hi*8
            G[(v + hi * 8) * G3 + ct * 16 + n16] = acc[v];
    }
    __syncthreads();

    // ---- phase 3: gate math, 16x128 elems, 8 per thread ----
    for (int idx = tid; idx < 16 * HDIM; idx += 256) {
        int r = idx >> 7, c = idx & 127;
        float ir = sGi[r * G3 + c]        + bih[c];
        float iz = sGi[r * G3 + 128 + c]  + bih[128 + c];
        float in_ = sGi[r * G3 + 256 + c] + bih[256 + c];
        float hr = sGh[r * G3 + c]        + bhh[c];
        float hz = sGh[r * G3 + 128 + c]  + bhh[128 + c];
        float hn = sGh[r * G3 + 256 + c]  + bhh[256 + c];
        float rg = 1.f / (1.f + expf(-(ir + hr)));
        float zg = 1.f / (1.f + expf(-(iz + hz)));
        float ng = tanhf(in_ + rg * hn);
        float h2 = (1.f - zg) * ng + zg * (float)sH[idx];
        bf16 h2b = (bf16)h2;
        Hst[row0 * HDIM + idx] = h2b;
        if (Xout) Xout[(row0 + r) * (size_t)xld + xoff + c] = h2b;
    }
}

// ---- helpers ---------------------------------------------------------------
__global__ void zero_ws_kernel(unsigned int* p, size_t nwords) {
    size_t i = (size_t)blockIdx.x * blockDim.x + threadIdx.x;
    if (i < nwords) p[i] = 0u;
}

// fp32 [rows][cols] -> bf16 [rows][colsPad] (zero pad)
__global__ void wconv_kernel(const float* __restrict__ W, bf16* __restrict__ Wb,
                             int cols, int colsPad) {
    int idx = blockIdx.x * 256 + threadIdx.x;      // rows*colsPad total
    int r = idx / colsPad, c = idx - r * colsPad;
    Wb[idx] = (c < cols) ? (bf16)W[(size_t)r * cols + c] : (bf16)0.f;
}

// node embedding: h = towers @ En_W^T + En_b  (16384 x 128, in=14)
__global__ void enc_kernel(const float* __restrict__ towers,
                           const float* __restrict__ W,
                           const float* __restrict__ b,
                           bf16* __restrict__ h) {
    int idx = blockIdx.x * 256 + threadIdx.x;      // NN*128
    int row = idx >> 7, c = idx & 127;
    float acc = b[c];
    #pragma unroll
    for (int t = 0; t < 14; ++t) acc += towers[(size_t)row * 14 + t] * W[c * 14 + t];
    h[idx] = (bf16)acc;
}

// x_edge[(n,i,j)] = concat(h[n,i], h[n,j])   (bf16 -> bf16 gather)
__global__ void build_x_edge_kernel(const bf16* __restrict__ h,
                                    bf16* __restrict__ x) {
    size_t idx = (size_t)blockIdx.x * 256 + threadIdx.x;   // NE*256
    size_t row = idx >> 8; int c = (int)(idx & 255);
    size_t ni = row >> 3;                                  // n*8+i
    size_t nj = (row >> 6) * 8 + (row & 7);                // n*8+j
    x[idx] = (c < 128) ? h[ni * 128 + c] : h[nj * 128 + (c - 128)];
}

// x_node[(n,i)] = concat(towers[n,i] (14), e[n,i,i+1] or 0 (128), pad to 160)
__global__ void build_x_node_kernel(const float* __restrict__ towers,
                                    const bf16* __restrict__ e,
                                    bf16* __restrict__ x) {
    size_t idx = (size_t)blockIdx.x * 256 + threadIdx.x;   // NN*160
    size_t row = idx / 160; int c = (int)(idx - row * 160);
    size_t n = row >> 3; int i = (int)(row & 7);
    bf16 v = (bf16)0.f;
    if (c < 14)        v = (bf16)towers[row * 14 + c];
    else if (c < 142) {
        int cc = c - 14;
        if (i < 7) v = e[(n * 64 + (size_t)i * 8 + (i + 1)) * 128 + cc];
    }
    x[idx] = v;
}

// x_glob[n] = concat(sum_i h[n,i], sum_ij e[n,i,j])
__global__ void build_x_glob_kernel(const bf16* __restrict__ h,
                                    const bf16* __restrict__ e,
                                    bf16* __restrict__ x) {
    int idx = blockIdx.x * 256 + threadIdx.x;              // NG*128
    int n = idx >> 7, c = idx & 127;
    float hs = 0.f, es = 0.f;
    #pragma unroll
    for (int i = 0; i < 8; ++i)  hs += (float)h[((size_t)n * 8 + i) * 128 + c];
    for (int p = 0; p < 64; ++p) es += (float)e[((size_t)n * 64 + p) * 128 + c];
    x[(size_t)n * 256 + c]       = (bf16)hs;
    x[(size_t)n * 256 + 128 + c] = (bf16)es;
}

// output MLP: 128->128->128->1, relu, relu, sigmoid. One block (128 thr) / row.
__global__ __launch_bounds__(128) void mlp_kernel(
    const bf16* __restrict__ g,
    const float* __restrict__ W0, const float* __restrict__ b0,
    const float* __restrict__ W1, const float* __restrict__ b1,
    const float* __restrict__ W2, const float* __restrict__ b2,
    float* __restrict__ out) {
    __shared__ float s0[128], s1[128];
    int r = blockIdx.x, t = threadIdx.x;
    s0[t] = (float)g[(size_t)r * 128 + t];
    __syncthreads();
    float acc = b0[t];
    for (int j = 0; j < 128; ++j) acc += W0[t * 128 + j] * s0[j];
    s1[t] = fmaxf(acc, 0.f);
    __syncthreads();
    acc = b1[t];
    for (int j = 0; j < 128; ++j) acc += W1[t * 128 + j] * s1[j];
    s0[t] = fmaxf(acc, 0.f);
    __syncthreads();
    if (t == 0) {
        float a = b2[0];
        for (int j = 0; j < 128; ++j) a += W2[j] * s0[j];
        out[r] = 1.f / (1.f + expf(-a));
    }
}

// ---------------------------------------------------------------------------
extern "C" void kernel_launch(void* const* d_in, const int* in_sizes, int n_in,
                              void* d_out, int out_size, void* d_ws, size_t ws_size,
                              hipStream_t stream) {
    (void)in_sizes; (void)n_in; (void)out_size; (void)ws_size;
    // input layout (setup_inputs dict order, leaves flattened):
    const float* towers = (const float*)d_in[0];
    // d_in[1] = k (fixed 3 in reference setup; device scalar unreadable in capture)
    const float* EnW = (const float*)d_in[2];
    const float* Enb = (const float*)d_in[3];
    // u_params: 4..11, m_params: 12..19, g_params: 20..27, O_params: 28..33

    // ---- workspace layout (all sizes 256B-aligned by construction) ----
    char* ws = (char*)d_ws;
    size_t off = 0;
    auto take = [&](size_t bytes) { char* p = ws + off; off += bytes; return p; };
    // zero-initialized state block (contiguous)
    bf16* e0 = (bf16*)take(NE * 128 * 2);
    bf16* e1 = (bf16*)take(NE * 128 * 2);
    bf16* h0 = (bf16*)take(NN * 128 * 2);
    bf16* h1 = (bf16*)take(NN * 128 * 2);
    bf16* g0 = (bf16*)take(NG * 128 * 2);
    bf16* g1 = (bf16*)take(NG * 128 * 2);
    size_t state_bytes = off;
    bf16* hn  = (bf16*)take(NN * 128 * 2);   // node representation h
    bf16* xe  = (bf16*)take(NE * 256 * 2);   // edge concat input
    bf16* xe2 = (bf16*)take(NE * 128 * 2);   // edge layer1 input
    bf16* xn  = (bf16*)take(NN * 160 * 2);   // node input (142 pad 160)
    bf16* xn2 = (bf16*)take(NN * 128 * 2);
    bf16* xg  = (bf16*)take(NG * 256 * 2);
    bf16* xg2 = (bf16*)take(NG * 128 * 2);
    // padded bf16 weights
    bf16* m0i = (bf16*)take(384 * 256 * 2); bf16* m0h = (bf16*)take(384 * 128 * 2);
    bf16* m1i = (bf16*)take(384 * 128 * 2); bf16* m1h = (bf16*)take(384 * 128 * 2);
    bf16* u0i = (bf16*)take(384 * 160 * 2); bf16* u0h = (bf16*)take(384 * 128 * 2);
    bf16* u1i = (bf16*)take(384 * 128 * 2); bf16* u1h = (bf16*)take(384 * 128 * 2);
    bf16* g0i = (bf16*)take(384 * 256 * 2); bf16* g0h = (bf16*)take(384 * 128 * 2);
    bf16* g1i = (bf16*)take(384 * 128 * 2); bf16* g1h = (bf16*)take(384 * 128 * 2);

    // ---- zero GRU states (harness poisons ws; must re-zero every call) ----
    size_t zwords = state_bytes / 4;
    zero_ws_kernel<<<(unsigned)((zwords + 255) / 256), 256, 0, stream>>>(
        (unsigned int*)d_ws, zwords);

    // ---- weight conversion fp32 -> padded bf16 ----
    auto wc = [&](int in_idx, bf16* dst, int cols, int colsPad) {
        wconv_kernel<<<(384 * colsPad) / 256, 256, 0, stream>>>(
            (const float*)d_in[in_idx], dst, cols, colsPad);
    };
    wc(12, m0i, 256, 256); wc(13, m0h, 128, 128);
    wc(16, m1i, 128, 128); wc(17, m1h, 128, 128);
    wc(4,  u0i, 142, 160); wc(5,  u0h, 128, 128);
    wc(8,  u1i, 128, 128); wc(9,  u1h, 128, 128);
    wc(20, g0i, 256, 256); wc(21, g0h, 128, 128);
    wc(24, g1i, 128, 128); wc(25, g1h, 128, 128);

    // ---- node embedding ----
    enc_kernel<<<(unsigned)(NN * 128 / 256), 256, 0, stream>>>(towers, EnW, Enb, hn);

    // ---- k = 3 message-passing iterations ----
    for (int it = 0; it < 3; ++it) {
        // edge GRU (M_gru), 131072 rows
        build_x_edge_kernel<<<(unsigned)(NE * 256 / 256), 256, 0, stream>>>(hn, xe);
        gru16_kernel<<<(unsigned)(NE / 16), 256, 0, stream>>>(
            xe, 256, e0, m0i, m0h, (const float*)d_in[14], (const float*)d_in[15],
            xe2, 128, 0);
        gru16_kernel<<<(unsigned)(NE / 16), 256, 0, stream>>>(
            xe2, 128, e1, m1i, m1h, (const float*)d_in[18], (const float*)d_in[19],
            (bf16*)nullptr, 0, 0);
        // node GRU (U_gru), 16384 rows; x = [towers | e[n,i,i+1]]
        build_x_node_kernel<<<(unsigned)(NN * 160 / 256), 256, 0, stream>>>(
            towers, e1, xn);
        gru16_kernel<<<(unsigned)(NN / 16), 256, 0, stream>>>(
            xn, 160, h0, u0i, u0h, (const float*)d_in[6], (const float*)d_in[7],
            xn2, 128, 0);
        gru16_kernel<<<(unsigned)(NN / 16), 256, 0, stream>>>(
            xn2, 128, h1, u1i, u1h, (const float*)d_in[10], (const float*)d_in[11],
            hn, 128, 0);                                    // new node rep
        // global GRU (G_gru), 2048 rows; x = [sum_i h | sum_ij e]
        build_x_glob_kernel<<<(unsigned)(NG * 128 / 256), 256, 0, stream>>>(hn, e1, xg);
        gru16_kernel<<<(unsigned)(NG / 16), 256, 0, stream>>>(
            xg, 256, g0, g0i, g0h, (const float*)d_in[22], (const float*)d_in[23],
            xg2, 128, 0);
        gru16_kernel<<<(unsigned)(NG / 16), 256, 0, stream>>>(
            xg2, 128, g1, g1i, g1h, (const float*)d_in[26], (const float*)d_in[27],
            (bf16*)nullptr, 0, 0);
    }

    // ---- output MLP + sigmoid ----
    mlp_kernel<<<(unsigned)NG, 128, 0, stream>>>(
        g1,
        (const float*)d_in[28], (const float*)d_in[29],
        (const float*)d_in[30], (const float*)d_in[31],
        (const float*)d_in[32], (const float*)d_in[33],
        (float*)d_out);
}
